// GATLayer_regular_65000035058127
// MI455X (gfx1250) — matrix operand order repaired and checked by reading, not compiled, not run on hardware
//
#include <hip/hip_runtime.h>
#include <hip/hip_bf16.h>

// ---------------------------------------------------------------------------
// GAT layer for MI455X (gfx1250, wave32).
//   x0_i = lrelu(x0 @ W1^T + b1) -> reduced immediately to a1 = x0_i @ a1_w + a1_b
//   x0_j = lrelu(x0 @ W2^T + b2) -> stored (L2-resident, 25.6 MB), a2 likewise
//   out  = x0 + segment_sum(sigmoid(a1[src]+a2[dst]) * x0_j[dst], src)
// GEMMs via v_wmma_f32_16x16x32_bf16 (f32 accumulate); edge stage is L2-bound
// gather + native f32 atomics.
// ---------------------------------------------------------------------------

typedef __attribute__((ext_vector_type(16))) __bf16 bf16x16;
typedef __attribute__((ext_vector_type(8)))  __bf16 bf16x8;
typedef __attribute__((ext_vector_type(8)))  float  f32x8;

#define D_FEAT 128

// ---------------------------------------------------------------------------
// prep: x0 -> bf16, W1/W2 -> bf16, out = x0 (residual init, every call)
// ---------------------------------------------------------------------------
__global__ void gat_prep_kernel(const float* __restrict__ x0,
                                const float* __restrict__ W1,
                                const float* __restrict__ W2,
                                __bf16* __restrict__ x0b,
                                __bf16* __restrict__ W1b,
                                __bf16* __restrict__ W2b,
                                float* __restrict__ out,
                                size_t nd)
{
    size_t stride = (size_t)gridDim.x * blockDim.x;
    for (size_t i = (size_t)blockIdx.x * blockDim.x + threadIdx.x; i < nd; i += stride) {
        float v = x0[i];
        x0b[i] = (__bf16)v;
        out[i] = v;                       // residual: out starts at x0
        if (i < (size_t)D_FEAT * D_FEAT) {
            W1b[i] = (__bf16)W1[i];
            W2b[i] = (__bf16)W2[i];
        }
    }
}

// ---------------------------------------------------------------------------
// gemm: one wave computes a 16-row x 128-col tile of lrelu(x0 @ W^T + b).
//  - STORE_X: write the activated tile to xj (W2 path)
//  - always reduces tile against aw to produce per-row score a_out
// Fragment layouts used here (bf16 16x16x32, contiguous k order for A and B):
//   A (16x32): lane m=lane&15, hi=lane>>4; elems [0..15]=A[m][kb+0..15],
//              kb = 32*ks + hi*16
//   B (32x16): B = W^T, so lane (n=lane&15) reads W[col=16c+n][32*ks+hi*16 .. +15]
//   C/D tile:  elem r of lane = C[r + 8*hi][lane&15]
// ---------------------------------------------------------------------------
template <bool STORE_X>
__global__ void gat_gemm_kernel(const __bf16* __restrict__ x0b,
                                const __bf16* __restrict__ Wb,
                                const float*  __restrict__ bias,
                                const float*  __restrict__ aw,
                                const float*  __restrict__ ab,
                                float* __restrict__ a_out,
                                float* __restrict__ xj,
                                int nTiles, int N)
{
    const int wave = threadIdx.x >> 5;
    const int lane = threadIdx.x & 31;
    const int tile = blockIdx.x * (blockDim.x >> 5) + wave;
    if (tile >= nTiles) return;           // wave-uniform: EXEC stays all-ones

    const int m  = lane & 15;
    const int hi = lane >> 4;
    const int rowBase = tile * 16;
    int row = rowBase + m;
    if (row >= N) row = N - 1;            // benign clamp (N % 16 == 0 here)

    f32x8 acc[8];
#pragma unroll
    for (int c = 0; c < 8; ++c) acc[c] = (f32x8)0.0f;

#pragma unroll
    for (int ks = 0; ks < 4; ++ks) {
        // ---- A fragment ----
        const __bf16* pA = x0b + (size_t)row * D_FEAT + ks * 32 + hi * 16;
        bf16x8 alo = *(const bf16x8*)(pA);
        bf16x8 ahi = *(const bf16x8*)(pA + 8);
        bf16x16 afrag;
#pragma unroll
        for (int i = 0; i < 8; ++i) { afrag[i] = alo[i]; afrag[i + 8] = ahi[i]; }

#pragma unroll
        for (int c = 0; c < 8; ++c) {
            // ---- B fragment: 16 contiguous bf16 of W row (16c + n) ----
            const __bf16* pB = Wb + (size_t)(c * 16 + m) * D_FEAT + ks * 32 + hi * 16;
            bf16x16 bfrag = *(const bf16x16*)pB;
            acc[c] = __builtin_amdgcn_wmma_f32_16x16x32_bf16(
                false, afrag, false, bfrag, (short)0, acc[c], false, false);
        }
    }

    // epilogue: bias + LeakyReLU(0.2), optional store, score reduction
    float bcol[8], awcol[8];
#pragma unroll
    for (int c = 0; c < 8; ++c) {
        bcol[c]  = bias[c * 16 + m];
        awcol[c] = aw[c * 16 + m];
    }

    float ssum[8];
#pragma unroll
    for (int r = 0; r < 8; ++r) ssum[r] = 0.0f;

#pragma unroll
    for (int c = 0; c < 8; ++c) {
#pragma unroll
        for (int r = 0; r < 8; ++r) {
            float v = acc[c][r] + bcol[c];
            v = (v >= 0.0f) ? v : 0.2f * v;
            if (STORE_X) {
                int orow = rowBase + hi * 8 + r;
                xj[(size_t)orow * D_FEAT + c * 16 + m] = v;
            }
            ssum[r] += v * awcol[c];
        }
    }

    // butterfly sum across the 16 lanes that share `hi` (cols of one row)
#pragma unroll
    for (int mask = 1; mask < 16; mask <<= 1) {
#pragma unroll
        for (int r = 0; r < 8; ++r)
            ssum[r] += __shfl_xor(ssum[r], mask, 32);
    }
    if (m == 0) {
        float abv = ab[0];
#pragma unroll
        for (int r = 0; r < 8; ++r)
            a_out[rowBase + hi * 8 + r] = ssum[r] + abv;
    }
}

// ---------------------------------------------------------------------------
// edge: one wave per edge (grid-stride). x0_j and out are L2-resident, so the
// gather + f32 atomic scatter runs at L2 bandwidth.
// ---------------------------------------------------------------------------
__global__ void gat_edge_kernel(const int* __restrict__ ei,
                                const float* __restrict__ a1,
                                const float* __restrict__ a2,
                                const float* __restrict__ xj,
                                float* __restrict__ out,
                                int E)
{
    const int lane = threadIdx.x & 31;
    const int wid  = (blockIdx.x * blockDim.x + threadIdx.x) >> 5;
    const int nw   = (gridDim.x * blockDim.x) >> 5;

    for (int e = wid; e < E; e += nw) {
        int s = ei[e];
        int d = ei[E + e];
        float z = a1[s] + a2[d];                  // broadcast loads
        float att = 1.0f / (1.0f + __expf(-z));   // sigmoid
        const float4* xr = (const float4*)(xj + (size_t)d * D_FEAT);
        float4 v = xr[lane];                      // 32 lanes x 16B = full row
        float* o = out + (size_t)s * D_FEAT + lane * 4;
        __hip_atomic_fetch_add(o + 0, att * v.x, __ATOMIC_RELAXED, __HIP_MEMORY_SCOPE_AGENT);
        __hip_atomic_fetch_add(o + 1, att * v.y, __ATOMIC_RELAXED, __HIP_MEMORY_SCOPE_AGENT);
        __hip_atomic_fetch_add(o + 2, att * v.z, __ATOMIC_RELAXED, __HIP_MEMORY_SCOPE_AGENT);
        __hip_atomic_fetch_add(o + 3, att * v.w, __ATOMIC_RELAXED, __HIP_MEMORY_SCOPE_AGENT);
    }
}

// ---------------------------------------------------------------------------
static inline size_t align256(size_t x) { return (x + 255) & ~(size_t)255; }

extern "C" void kernel_launch(void* const* d_in, const int* in_sizes, int n_in,
                              void* d_out, int out_size, void* d_ws, size_t ws_size,
                              hipStream_t stream)
{
    const float* x0  = (const float*)d_in[0];
    // d_in[1] = x1 : unused by the reference
    const int*   ei  = (const int*)d_in[2];
    const float* W1  = (const float*)d_in[3];
    const float* b1  = (const float*)d_in[4];
    const float* W2  = (const float*)d_in[5];
    const float* b2  = (const float*)d_in[6];
    const float* a1w = (const float*)d_in[7];
    const float* a1b = (const float*)d_in[8];
    const float* a2w = (const float*)d_in[9];
    const float* a2b = (const float*)d_in[10];
    float* out = (float*)d_out;

    const int N = in_sizes[0] / D_FEAT;
    const int E = in_sizes[2] / 2;
    const size_t nd = (size_t)N * D_FEAT;

    // workspace carve-out (~39 MB)
    char* ws = (char*)d_ws;
    size_t off = 0;
    __bf16* x0b = (__bf16*)(ws + off); off = align256(off + nd * sizeof(__bf16));
    __bf16* W1b = (__bf16*)(ws + off); off = align256(off + (size_t)D_FEAT * D_FEAT * sizeof(__bf16));
    __bf16* W2b = (__bf16*)(ws + off); off = align256(off + (size_t)D_FEAT * D_FEAT * sizeof(__bf16));
    float*  xj  = (float*)(ws + off);  off = align256(off + nd * sizeof(float));
    float*  a1  = (float*)(ws + off);  off = align256(off + (size_t)N * sizeof(float));
    float*  a2  = (float*)(ws + off);  off = align256(off + (size_t)N * sizeof(float));
    (void)ws_size; (void)n_in; (void)out_size;

    // 1) convert + residual init
    gat_prep_kernel<<<8192, 256, 0, stream>>>(x0, W1, W2, x0b, W1b, W2b, out, nd);

    // 2) projections: 1 wave per 16-row tile, 8 waves per block
    const int nTiles = (N + 15) / 16;
    const int gemmBlocks = (nTiles + 7) / 8;
    gat_gemm_kernel<false><<<gemmBlocks, 256, 0, stream>>>(
        x0b, W1b, b1, a1w, a1b, a1, nullptr, nTiles, N);
    gat_gemm_kernel<true><<<gemmBlocks, 256, 0, stream>>>(
        x0b, W2b, b2, a2w, a2b, a2, xj, nTiles, N);

    // 3) edge gather / attention / atomic scatter
    gat_edge_kernel<<<2048, 256, 0, stream>>>(ei, a1, a2, xj, out, E);
}
